// CliffordNetBlock_1047972020604
// MI455X (gfx1250) — matrix operand order repaired and checked by reading, not compiled, not run on hardware
//
#include <hip/hip_runtime.h>
#include <cstdint>
#include <cstddef>

typedef __attribute__((ext_vector_type(16))) _Float16 v16h;
typedef __attribute__((ext_vector_type(8)))  float    v8f;

#define Dc 256
#define NPOS (8 * 96 * 96)          // 73728 positions
#define POS_PER_BATCH (96 * 96)     // 9216
#define TM 64                       // M-tile of main kernel
#define NBLK_MAIN (NPOS / TM)       // 1152
#define NTILE1 (NPOS / 16)          // 4608 (LN/sum pass tiles)
#define CHUNK_TILES 48              // tiles per stage-A reduction block
#define NCHUNK (POS_PER_BATCH / 16 / CHUNK_TILES) // 12 chunks per batch
#define LNEPS 1e-5f

// ---------------------------------------------------------------------------
// WMMA helpers (wave32, 16x16x32 f16 -> f32)
// ---------------------------------------------------------------------------
__device__ __forceinline__ v8f wmma_f16(v16h a, v16h b, v8f c) {
  // (neg_a, A, neg_b, B, c_mod, C, reuse_a, reuse_b)
  return __builtin_amdgcn_wmma_f32_16x16x32_f16(false, a, false, b, (short)0, c,
                                                false, false);
}

// Load a 16x32 f16 A-fragment from LDS (row-major, `stride` halfs per row).
// ISA layout: lanes 0-15 hold M=0..15 with K blocks [0..7, 16..23];
//             lanes 16-31 hold M=0..15 with K blocks [8..15, 24..31].
__device__ __forceinline__ v16h load_a16(const _Float16* __restrict__ base,
                                         int stride, int lane) {
  const int m  = lane & 15;
  const int kb = (lane & 16) ? 8 : 0;
  const _Float16* r = base + m * stride + kb;
  v16h a;
#pragma unroll
  for (int j = 0; j < 8; ++j) a[j] = r[j];          // K = kb + 0..7
#pragma unroll
  for (int j = 0; j < 8; ++j) a[8 + j] = r[16 + j]; // K = kb + 16..23
  return a;
}

// ---------------------------------------------------------------------------
// Kernel 0: pack fp32 weights into f16 WMMA B-fragment layout.
// B-fragment (32x16 f16): lane L holds column n = L%16, K = (L<16?0:8)+(j&7)+(j<8?0:16),
// 16 contiguous halfs per lane (one 32B load in the GEMM loop).
// Fragment f = kt*16 + nt occupies 512 halfs.
// ---------------------------------------------------------------------------
#define PROJ_HALVES (32 * 16 * 512) // K=1024 -> 32 ktiles
#define GATE_HALVES (16 * 16 * 512) // K=512  -> 16 ktiles

__global__ void pack_weights(const float* __restrict__ pk,
                             const float* __restrict__ gk,
                             _Float16* __restrict__ pp,
                             _Float16* __restrict__ pg) {
  int idx = blockIdx.x * 256 + threadIdx.x;
  if (idx < PROJ_HALVES) {
    int f = idx >> 9, rem = idx & 511;
    int lane = rem >> 4, j = rem & 15;
    int kt = f >> 4, nt = f & 15;
    int n = (nt << 4) + (lane & 15);
    int k = (kt << 5) + ((lane & 16) ? 8 : 0) + (j & 7) + ((j & 8) ? 16 : 0);
    pp[idx] = (_Float16)pk[k * Dc + n];
  } else if (idx < PROJ_HALVES + GATE_HALVES) {
    int i2 = idx - PROJ_HALVES;
    int f = i2 >> 9, rem = i2 & 511;
    int lane = rem >> 4, j = rem & 15;
    int kt = f >> 4, nt = f & 15;
    int n = (nt << 4) + (lane & 15);
    int k = (kt << 5) + ((lane & 16) ? 8 : 0) + (j & 7) + ((j & 8) ? 16 : 0);
    pg[i2] = (_Float16)gk[k * Dc + n];
  }
}

// ---------------------------------------------------------------------------
// LayerNorm of one row (256 ch) by one wave; lane covers 8 contiguous chans.
// ---------------------------------------------------------------------------
__device__ __forceinline__ void ln_row(const float* __restrict__ xrow,
                                       const float* __restrict__ lng,
                                       const float* __restrict__ lnb,
                                       int lane, float* __restrict__ dst) {
  const float4* xp = (const float4*)(xrow + lane * 8);
  float4 a = xp[0], b = xp[1];
  float v[8] = {a.x, a.y, a.z, a.w, b.x, b.y, b.z, b.w};
  float s = 0.f, q = 0.f;
#pragma unroll
  for (int i = 0; i < 8; ++i) { s += v[i]; q += v[i] * v[i]; }
#pragma unroll
  for (int off = 16; off > 0; off >>= 1) {
    s += __shfl_xor(s, off, 32);
    q += __shfl_xor(q, off, 32);
  }
  float mu  = s * (1.0f / 256.0f);
  float var = q * (1.0f / 256.0f) - mu * mu;
  float inv = rsqrtf(var + LNEPS);
#pragma unroll
  for (int i = 0; i < 8; ++i) {
    int c = lane * 8 + i;
    dst[c] = (v[i] - mu) * inv * lng[c] + lnb[c];
  }
}

// ---------------------------------------------------------------------------
// Kernel 1: LN + per-16-row channel sums (for the spatial mean).
// ---------------------------------------------------------------------------
__global__ __launch_bounds__(512) void ln_tilesum(const float* __restrict__ x,
                                                  const float* __restrict__ lng,
                                                  const float* __restrict__ lnb,
                                                  float* __restrict__ tileSums) {
  __shared__ float hn[16 * Dc];
  const int tid = threadIdx.x, wave = tid >> 5, lane = tid & 31;
  const int tile = blockIdx.x;
  const int pos = tile * 16 + wave;
  ln_row(x + (size_t)pos * Dc, lng, lnb, lane, hn + wave * Dc);
  __syncthreads();
  if (tid < Dc) {
    float s = 0.f;
#pragma unroll
    for (int r = 0; r < 16; ++r) s += hn[r * Dc + tid];
    tileSums[(size_t)tile * Dc + tid] = s;
  }
}

// ---------------------------------------------------------------------------
// Kernel 2a/2b: two-stage reduction of tile sums -> per-(batch,ch) mean.
// ---------------------------------------------------------------------------
__global__ void reduce_mean_a(const float* __restrict__ tileSums,
                              float* __restrict__ partial) {
  const int b = blockIdx.x / NCHUNK, ch = blockIdx.x % NCHUNK;
  const int d = threadIdx.x;
  const int TPB = POS_PER_BATCH / 16; // 576 tiles per batch
  float s = 0.f;
  const float* p =
      tileSums + ((size_t)b * TPB + (size_t)ch * CHUNK_TILES) * Dc + d;
#pragma unroll 4
  for (int t = 0; t < CHUNK_TILES; ++t) s += p[(size_t)t * Dc];
  partial[(size_t)blockIdx.x * Dc + d] = s;
}

__global__ void reduce_mean_b(const float* __restrict__ partial,
                              float* __restrict__ meanBD) {
  const int b = blockIdx.x, d = threadIdx.x;
  float s = 0.f;
#pragma unroll
  for (int c = 0; c < NCHUNK; ++c)
    s += partial[((size_t)b * NCHUNK + c) * Dc + d];
  meanBD[b * Dc + d] = s * (1.0f / (float)POS_PER_BATCH);
}

// ---------------------------------------------------------------------------
// Kernel 3: fused LN -> Clifford feats -> proj GEMM -> gate GEMM -> epilogue.
// Block: 256 threads = 8 waves; tile = 64 positions x 256 channels.
// Wave w owns output N-tiles {2w, 2w+1} across 4 row-blocks of 16.
// (8 waves x 2 N-tiles halves the LDS A-fragment broadcast traffic vs 16x1.)
// ---------------------------------------------------------------------------
__global__ __launch_bounds__(256, 1) void clifford_main(
    const float* __restrict__ x, const float* __restrict__ lng,
    const float* __restrict__ lnb, const float* __restrict__ projBias,
    const float* __restrict__ gateBias, const float* __restrict__ gamma,
    const float* __restrict__ meanBD, const _Float16* __restrict__ packProj,
    const _Float16* __restrict__ packGate, float* __restrict__ out) {
  // LDS plan: [0,64K)   hnF32  [64][256] fp32
  //           [64K,192K) feats [64][1024] f16   (overlaid by gateA [64][512])
  __shared__ __align__(32) unsigned char smem[196608];
  float*    hnF32 = (float*)smem;
  _Float16* feats = (_Float16*)(smem + 65536);
  _Float16* gateA = (_Float16*)(smem + 65536); // reuse after proj GEMM

  const int tid  = threadIdx.x;
  const int wave = tid >> 5;
  const int lane = tid & 31;
  const int tileBase = blockIdx.x * TM;
  const int batch = blockIdx.x / (POS_PER_BATCH / TM); // 144 blocks / batch

  // ---- stage 1: LayerNorm (each wave: rows 8w..8w+7) ----
#pragma unroll
  for (int rr = 0; rr < 8; ++rr) {
    int row = wave * 8 + rr;
    ln_row(x + (size_t)(tileBase + row) * Dc, lng, lnb, lane, hnF32 + row * Dc);
  }
  __syncthreads();

  // ---- stage 2: Clifford features (f16 into LDS) ----
  // feats[row][blk*256+c]: blk0=dot(s=1) blk1=wedge(s=1) blk2=dot(s=2) blk3=wedge(s=2)
  const float* mrow = meanBD + batch * Dc;
  for (int i = tid; i < TM * 1024; i += 256) {
    int row = i >> 10;
    int cp  = i & 1023;
    int blk = cp >> 8;
    int c   = cp & 255;
    int sft = (blk >> 1) + 1;           // SHIFTS = (1,2)
    int cs  = (c + sft) & 255;
    float det  = hnF32[row * Dc + c];
    float dets = hnF32[row * Dc + cs];
    float ctx  = det - mrow[c];
    float ctxs = dets - mrow[cs];
    float p = det * ctxs;
    float val = (blk & 1) ? (p - ctx * dets)          // wedge
                          : (p / (1.0f + expf(-p)));  // silu(dot)
    feats[i] = (_Float16)val;
  }
  __syncthreads();

  // ---- stage 3: proj GEMM  D=feats[64,1024] @ pk[1024,256] ----
  const int nt0 = wave * 2, nt1 = wave * 2 + 1;
  const int ncol0 = nt0 * 16 + (lane & 15);
  const int ncol1 = nt1 * 16 + (lane & 15);
  v8f zero = {0.f, 0.f, 0.f, 0.f, 0.f, 0.f, 0.f, 0.f};
  v8f accP[4][2] = {{zero, zero}, {zero, zero}, {zero, zero}, {zero, zero}};
  for (int kt = 0; kt < 32; ++kt) {
    v16h b0 =
        *(const v16h*)(packProj + ((size_t)(kt * 16 + nt0) * 512) + lane * 16);
    v16h b1 =
        *(const v16h*)(packProj + ((size_t)(kt * 16 + nt1) * 512) + lane * 16);
#pragma unroll
    for (int m = 0; m < 4; ++m) {
      v16h afrag = load_a16(feats + (m * 16) * 1024 + kt * 32, 1024, lane);
      accP[m][0] = wmma_f16(afrag, b0, accP[m][0]);
      accP[m][1] = wmma_f16(afrag, b1, accP[m][1]);
    }
  }
  float pb0 = projBias[ncol0], pb1 = projBias[ncol1];
#pragma unroll
  for (int m = 0; m < 4; ++m)
#pragma unroll
    for (int r = 0; r < 8; ++r) {
      accP[m][0][r] += pb0;
      accP[m][1][r] += pb1;
    }
  __syncthreads(); // all waves done reading feats; safe to overlay gateA

  // ---- stage 4: build gate A = [h_norm | g_feat] in f16 ----
  for (int i = tid; i < TM * Dc; i += 256) {
    int row = i >> 8, c = i & 255;
    gateA[row * 512 + c] = (_Float16)hnF32[row * Dc + c];
  }
#pragma unroll
  for (int m = 0; m < 4; ++m)
#pragma unroll
    for (int r = 0; r < 8; ++r) {
      int row = m * 16 + ((lane & 16) ? r + 8 : r); // C-fragment row map
      gateA[row * 512 + 256 + ncol0] = (_Float16)accP[m][0][r];
      gateA[row * 512 + 256 + ncol1] = (_Float16)accP[m][1][r];
    }
  __syncthreads();

  // ---- stage 5: gate GEMM  [64,512] @ gk[512,256] ----
  v8f accG[4][2] = {{zero, zero}, {zero, zero}, {zero, zero}, {zero, zero}};
  for (int kt = 0; kt < 16; ++kt) {
    v16h b0 =
        *(const v16h*)(packGate + ((size_t)(kt * 16 + nt0) * 512) + lane * 16);
    v16h b1 =
        *(const v16h*)(packGate + ((size_t)(kt * 16 + nt1) * 512) + lane * 16);
#pragma unroll
    for (int m = 0; m < 4; ++m) {
      v16h afrag = load_a16(gateA + (m * 16) * 512 + kt * 32, 512, lane);
      accG[m][0] = wmma_f16(afrag, b0, accG[m][0]);
      accG[m][1] = wmma_f16(afrag, b1, accG[m][1]);
    }
  }

  // ---- epilogue: out = x + (silu(hn) + sigmoid(gate)*g_feat) * gamma ----
  float gb0 = gateBias[ncol0], gb1 = gateBias[ncol1];
  float gm0 = gamma[ncol0], gm1 = gamma[ncol1];
#pragma unroll
  for (int m = 0; m < 4; ++m) {
#pragma unroll
    for (int r = 0; r < 8; ++r) {
      int row = m * 16 + ((lane & 16) ? r + 8 : r);
      size_t pos = (size_t)(tileBase + row);
      {
        float hn  = hnF32[row * Dc + ncol0];
        float g   = accP[m][0][r];
        float al  = 1.0f / (1.0f + expf(-(accG[m][0][r] + gb0)));
        float sil = hn / (1.0f + expf(-hn));
        out[pos * Dc + ncol0] = x[pos * Dc + ncol0] + (sil + al * g) * gm0;
      }
      {
        float hn  = hnF32[row * Dc + ncol1];
        float g   = accP[m][1][r];
        float al  = 1.0f / (1.0f + expf(-(accG[m][1][r] + gb1)));
        float sil = hn / (1.0f + expf(-hn));
        out[pos * Dc + ncol1] = x[pos * Dc + ncol1] + (sil + al * g) * gm1;
      }
    }
  }
}

// ---------------------------------------------------------------------------
// Host launcher
// ---------------------------------------------------------------------------
extern "C" void kernel_launch(void* const* d_in, const int* in_sizes, int n_in,
                              void* d_out, int out_size, void* d_ws,
                              size_t ws_size, hipStream_t stream) {
  (void)in_sizes; (void)n_in; (void)out_size; (void)ws_size;
  const float* x   = (const float*)d_in[0];
  const float* lng = (const float*)d_in[1];
  const float* lnb = (const float*)d_in[2];
  const float* pk  = (const float*)d_in[3];
  const float* pb  = (const float*)d_in[4];
  const float* gk  = (const float*)d_in[5];
  const float* gb  = (const float*)d_in[6];
  const float* gm  = (const float*)d_in[7];
  float* out = (float*)d_out;

  // workspace layout (all regions fully overwritten each launch)
  char* ws = (char*)d_ws;
  _Float16* pp       = (_Float16*)(ws);                           // 512 KB
  _Float16* pg       = (_Float16*)(ws + 524288);                  // 256 KB
  float*    tileSums = (float*)(ws + 524288 + 262144);            // 4.5 MB
  float*    partial  = (float*)(ws + 524288 + 262144 + 4718592);  // 96 KB
  float*    meanBD   = (float*)(ws + 524288 + 262144 + 4718592 + 98304); // 8 KB

  pack_weights<<<(PROJ_HALVES + GATE_HALVES) / 256, 256, 0, stream>>>(pk, gk,
                                                                      pp, pg);
  ln_tilesum<<<NTILE1, 512, 0, stream>>>(x, lng, lnb, tileSums);
  reduce_mean_a<<<8 * NCHUNK, 256, 0, stream>>>(tileSums, partial);
  reduce_mean_b<<<8, 256, 0, stream>>>(partial, meanBD);
  clifford_main<<<NBLK_MAIN, 256, 0, stream>>>(x, lng, lnb, pb, gb, gm, meanBD,
                                               pp, pg, out);
}
